// TriSourceQueryRouter_40699110097151
// MI455X (gfx1250) — matrix operand order repaired and checked by reading, not compiled, not run on hardware
//
#include <hip/hip_runtime.h>
#include <hip/hip_bf16.h>
#include <math.h>

typedef __attribute__((ext_vector_type(16))) _Float16 v16h;
typedef __attribute__((ext_vector_type(8)))  float    v8f;

#define BB 8
#define NL 20000
#define NP 2000
#define NG 1000
#define NITEMS 23000
#define DD 256
#define KSEL 900
#define TILES_PER_B 1438   // ceil(23000/16)
#define MLP_BLOCKS 512
#define MLP_WAVES (MLP_BLOCKS*8)
#define LPITCH 272         // LDS row pitch in halves: 544B = 136 dw = 8 mod 64 banks
#define LDS_BYTES (256 * LPITCH * 2)

// output offsets (floats)
#define O_GQ  0u
#define O_GR  1843200u
#define O_GS  1864800u
#define O_SRC 1872000u
#define O_GL  1879200u

__device__ __forceinline__ unsigned f2key(float f) {
    unsigned u = __float_as_uint(f);
    return (u & 0x80000000u) ? ~u : (u | 0x80000000u); // order-preserving
}

// ---------------------------------------------------------------------------
// K0: w1[0:256,:] (f32, [k][n]) -> w1t (f16, [n][k])
// ---------------------------------------------------------------------------
__global__ __launch_bounds__(256) void conv_w1(const float* __restrict__ w1,
                                               _Float16* __restrict__ w1t) {
    int gid = blockIdx.x * 256 + threadIdx.x;   // 65536 total
    int n = gid >> 8, k = gid & 255;
    w1t[gid] = (_Float16)w1[k * 256 + n];
}

// ---------------------------------------------------------------------------
// K1: per (batch,source) min/max of transformed valid scores
// ---------------------------------------------------------------------------
__global__ __launch_bounds__(256) void norm_reduce(const float* __restrict__ lsc,
                                                   const float* __restrict__ psc,
                                                   const float* __restrict__ gsc,
                                                   float* __restrict__ normP) {
    int bs = blockIdx.x, b = bs / 3, s = bs % 3;
    int n = (s == 0) ? NL : ((s == 1) ? NP : NG);
    float mn = INFINITY, mx = -INFINITY;
    for (int i = threadIdx.x; i < n; i += 256) {
        float t; bool valid;
        if (s == 0)      { float r = lsc[b * NL + i]; valid = r > 0.f; t = log1pf(r); }
        else if (s == 1) { float r = psc[b * NP + i]; valid = r > 0.f; t = r; }
        else             { t = gsc[b * NG + i]; valid = true; }
        if (valid) { mn = fminf(mn, t); mx = fmaxf(mx, t); }
    }
    __shared__ float smn[256], smx[256];
    int tid = threadIdx.x;
    smn[tid] = mn; smx[tid] = mx;
    __syncthreads();
    for (int o = 128; o > 0; o >>= 1) {
        if (tid < o) { smn[tid] = fminf(smn[tid], smn[tid + o]);
                       smx[tid] = fmaxf(smx[tid], smx[tid + o]); }
        __syncthreads();
    }
    if (tid == 0) {
        float mnv = smn[0], mxv = smx[0];
        float rng = mxv - mnv;
        int small = (fabsf(rng) < 1e-6f) ? 1 : 0;      // NaN -> false, like jnp
        float mn_safe = isfinite(mnv) ? mnv : 0.f;
        float rng_safe = (small || !isfinite(rng)) ? 1.f : rng;
        normP[bs * 3 + 0] = mn_safe;
        normP[bs * 3 + 1] = rng_safe;
        normP[bs * 3 + 2] = (float)small;
    }
}

// ---------------------------------------------------------------------------
// K2: elementwise normalized score -> all_scores[b][item]
// ---------------------------------------------------------------------------
__global__ __launch_bounds__(256) void norm_apply(const float* __restrict__ lsc,
                                                  const float* __restrict__ psc,
                                                  const float* __restrict__ gsc,
                                                  const float* __restrict__ normP,
                                                  float* __restrict__ scores) {
    int gid = blockIdx.x * 256 + threadIdx.x;
    if (gid >= BB * NITEMS) return;
    int b = gid / NITEMS, item = gid % NITEMS;
    int s; float t; bool valid;
    if (item < NL)            { s = 0; float r = lsc[b * NL + item]; valid = r > 0.f; t = log1pf(r); }
    else if (item < NL + NP)  { s = 1; float r = psc[b * NP + (item - NL)]; valid = r > 0.f; t = r; }
    else                      { s = 2; t = gsc[b * NG + (item - NL - NP)]; valid = true; }
    int p = (b * 3 + s) * 3;
    float v = (t - normP[p]) / normP[p + 1];
    if (normP[p + 2] != 0.f) v = 1.f;
    if (!valid) v = 0.f;
    scores[gid] = v;
}

// ---------------------------------------------------------------------------
// K3: MLP via WMMA f16: h = relu([q+emb, s] @ w1 + b1); logit = h @ w2 + b2
// ---------------------------------------------------------------------------
__global__ __launch_bounds__(256) void mlp_wmma(
        const float* __restrict__ lq, const float* __restrict__ pq,
        const float* __restrict__ gq, const float* __restrict__ se,
        const float* __restrict__ w1, const float* __restrict__ b1,
        const float* __restrict__ w2, const float* __restrict__ b2,
        const _Float16* __restrict__ w1t, const float* __restrict__ scores,
        float* __restrict__ keepL, float* __restrict__ keepS) {
    extern __shared__ _Float16 w1s[];          // 256 rows x 272 halves (padded)
    {   // stage w1^T into LDS, row pitch padded to kill bank conflicts
#if __has_builtin(__builtin_amdgcn_global_load_async_to_lds_b128)
        typedef __attribute__((__vector_size__(4 * sizeof(int)))) int vi4;
        typedef __attribute__((address_space(1))) vi4 gvi4;
        typedef __attribute__((address_space(3))) vi4 lvi4;
        for (int i = threadIdx.x; i < 8192; i += 256) {
            int row = i >> 5, c = i & 31;                 // 32 x 16B chunks/row
            const _Float16* sp = w1t + row * 256 + c * 8;
            _Float16* dp = w1s + row * LPITCH + c * 8;
            __builtin_amdgcn_global_load_async_to_lds_b128(
                (gvi4*)(vi4*)(void*)sp, (lvi4*)(vi4*)(void*)dp, 0, 0);
        }
#if __has_builtin(__builtin_amdgcn_s_wait_asynccnt)
        __builtin_amdgcn_s_wait_asynccnt(0);
#else
        asm volatile("s_wait_asynccnt 0" ::: "memory");
#endif
#else
        for (int i = threadIdx.x; i < 8192; i += 256) {
            int row = i >> 5, c = i & 31;
            *(float4*)(w1s + row * LPITCH + c * 8) =
                *(const float4*)(w1t + row * 256 + c * 8);
        }
#endif
    }
    __syncthreads();

    const int lane = threadIdx.x & 31;
    const int l  = lane & 15;
    const int hi = lane >> 4;
    const int waveId = blockIdx.x * 8 + (threadIdx.x >> 5);
    const float* w1last = w1 + 256 * 256;      // row 256 of w1 (score column)
    const float b2v = b2[0];

    // epilogue constants (invariant across tiles): n = nt*16 + l
    float b1v[16], wlv[16], w2v[16];
    #pragma unroll
    for (int nt = 0; nt < 16; ++nt) {
        int n = nt * 16 + l;
        b1v[nt] = b1[n]; wlv[nt] = w1last[n]; w2v[nt] = w2[n];
    }

    for (int t = waveId; t < BB * TILES_PER_B; t += MLP_WAVES) {
        int b = t / TILES_PER_B;
        int base = (t % TILES_PER_B) * 16;

        // this lane supplies A row M = l (both half-waves load row l, K split)
        int it = base + l; if (it > NITEMS - 1) it = NITEMS - 1;
        const float* qrow; int src;
        if (it < NL)           { qrow = lq + ((size_t)b * NL + it) * DD;              src = 0; }
        else if (it < NL + NP) { qrow = pq + ((size_t)b * NP + (it - NL)) * DD;       src = 1; }
        else                   { qrow = gq + ((size_t)b * NG + (it - NL - NP)) * DD;  src = 2; }
        const float* erow = se + src * DD;

        // prefetch next tile's A rows behind this tile's 128 WMMAs
        int t2 = t + MLP_WAVES;
        if (t2 < BB * TILES_PER_B) {
            int b2i = t2 / TILES_PER_B;
            int it2 = (t2 % TILES_PER_B) * 16 + l;
            if (it2 > NITEMS - 1) it2 = NITEMS - 1;
            const float* q2;
            if (it2 < NL)           q2 = lq + ((size_t)b2i * NL + it2) * DD;
            else if (it2 < NL + NP) q2 = pq + ((size_t)b2i * NP + (it2 - NL)) * DD;
            else                    q2 = gq + ((size_t)b2i * NG + (it2 - NL - NP)) * DD;
            __builtin_prefetch(q2, 0, 1);
            __builtin_prefetch(q2 + 128, 0, 1);
        }

        float sv = scores[b * NITEMS + it];
        float sm[8];
        #pragma unroll
        for (int r = 0; r < 8; ++r) sm[r] = __shfl(sv, r + 8 * hi, 32);

        // A fragments for all 8 K-tiles (ISA 16-bit A 16x32 layout)
        v16h A[8];
        #pragma unroll
        for (int kt = 0; kt < 8; ++kt) {
            int kA = kt * 32 + hi * 8;
            int kB = kt * 32 + 16 + hi * 8;
            float4 q0 = *(const float4*)(qrow + kA);
            float4 q1 = *(const float4*)(qrow + kA + 4);
            float4 e0 = *(const float4*)(erow + kA);
            float4 e1 = *(const float4*)(erow + kA + 4);
            float4 q2 = *(const float4*)(qrow + kB);
            float4 q3 = *(const float4*)(qrow + kB + 4);
            float4 e2 = *(const float4*)(erow + kB);
            float4 e3 = *(const float4*)(erow + kB + 4);
            v16h a;
            a[0]  = (_Float16)(q0.x + e0.x); a[1]  = (_Float16)(q0.y + e0.y);
            a[2]  = (_Float16)(q0.z + e0.z); a[3]  = (_Float16)(q0.w + e0.w);
            a[4]  = (_Float16)(q1.x + e1.x); a[5]  = (_Float16)(q1.y + e1.y);
            a[6]  = (_Float16)(q1.z + e1.z); a[7]  = (_Float16)(q1.w + e1.w);
            a[8]  = (_Float16)(q2.x + e2.x); a[9]  = (_Float16)(q2.y + e2.y);
            a[10] = (_Float16)(q2.z + e2.z); a[11] = (_Float16)(q2.w + e2.w);
            a[12] = (_Float16)(q3.x + e3.x); a[13] = (_Float16)(q3.y + e3.y);
            a[14] = (_Float16)(q3.z + e3.z); a[15] = (_Float16)(q3.w + e3.w);
            A[kt] = a;
        }

        float plog[8];
        #pragma unroll
        for (int r = 0; r < 8; ++r) plog[r] = 0.f;

        for (int nt = 0; nt < 16; ++nt) {
            v8f acc = {0.f, 0.f, 0.f, 0.f, 0.f, 0.f, 0.f, 0.f};
            const _Float16* bp = w1s + (nt * 16 + l) * LPITCH + (hi << 3);
            #pragma unroll
            for (int kt = 0; kt < 8; ++kt) {
                typedef __attribute__((ext_vector_type(8))) _Float16 v8h_;
                v8h_ b0 = *(const v8h_*)(bp + kt * 32);
                v8h_ b1c = *(const v8h_*)(bp + kt * 32 + 16);
                v16h Bf = __builtin_shufflevector(b0, b1c, 0,1,2,3,4,5,6,7,8,9,10,11,12,13,14,15);
                acc = __builtin_amdgcn_wmma_f32_16x16x32_f16(
                          false, A[kt], false, Bf, (short)0, acc, false, false);
            }
            float b1n = b1v[nt], wl = wlv[nt], w2n = w2v[nt];
            #pragma unroll
            for (int r = 0; r < 8; ++r) {
                float h = acc[r] + sm[r] * wl + b1n;     // add score col + bias
                h = h > 0.f ? h : 0.f;                   // relu
                plog[r] += h * w2n;                      // fold h @ w2
            }
        }
        #pragma unroll
        for (int r = 0; r < 8; ++r) {
            plog[r] += __shfl_xor(plog[r], 1, 32);
            plog[r] += __shfl_xor(plog[r], 2, 32);
            plog[r] += __shfl_xor(plog[r], 4, 32);
            plog[r] += __shfl_xor(plog[r], 8, 32);
        }
        if (l == 0) {   // lanes 0 (M=0..7) and 16 (M=8..15)
            #pragma unroll
            for (int r = 0; r < 8; ++r) {
                int item = base + r + 8 * hi;
                if (item < NITEMS) {
                    int row = b * NITEMS + item;
                    float lg = plog[r] + b2v;
                    keepL[row] = lg;
                    keepS[row] = lg + sm[r];
                }
            }
        }
    }
}

// ---------------------------------------------------------------------------
// K4: per (batch,source) radix-select of quota-th largest keep_score (valid)
// ---------------------------------------------------------------------------
__global__ __launch_bounds__(256) void tier_select(const float* __restrict__ lsc,
                                                   const float* __restrict__ psc,
                                                   const float* __restrict__ keepS,
                                                   int* __restrict__ tsel) {
    int bs = blockIdx.x, b = bs / 3, s = bs % 3;
    int off, n, quota;
    if (s == 0)      { off = 0;       n = NL; quota = 600; }
    else if (s == 1) { off = NL;      n = NP; quota = 200; }
    else             { off = NL + NP; n = NG; quota = 100; }
    __shared__ unsigned hist[256];
    __shared__ unsigned prefix_s;
    __shared__ int kk_s, allsel_s;
    if (threadIdx.x == 0) { prefix_s = 0; kk_s = quota; allsel_s = 0; }
    __syncthreads();
    for (int byt = 3; byt >= 0; --byt) {
        hist[threadIdx.x] = 0;
        __syncthreads();
        unsigned pre = prefix_s;
        for (int i = threadIdx.x; i < n; i += 256) {
            bool valid;
            if (s == 0)      valid = lsc[b * NL + i] > 0.f;
            else if (s == 1) valid = psc[b * NP + i] > 0.f;
            else             valid = true;
            if (!valid) continue;
            unsigned key = f2key(keepS[b * NITEMS + off + i]);
            if (byt == 3 || ((key ^ pre) >> ((byt + 1) * 8)) == 0)
                atomicAdd(&hist[(key >> (byt * 8)) & 255], 1u);
        }
        __syncthreads();
        if (threadIdx.x == 0) {
            int kk = kk_s, found = -1;
            for (int bin = 255; bin >= 0; --bin) {
                int c = (int)hist[bin];
                if (c >= kk) { found = bin; break; }
                kk -= c;
            }
            if (found < 0) allsel_s = 1;   // fewer valid than quota
            else { prefix_s = pre | ((unsigned)found << (byt * 8)); kk_s = kk; }
        }
        __syncthreads();
        if (allsel_s) break;
    }
    if (threadIdx.x == 0) {
        int p = bs * 4;
        if (allsel_s) { tsel[p] = 0; tsel[p + 1] = 0; }
        else          { tsel[p] = (int)prefix_s; tsel[p + 1] = kk_s; }
        tsel[p + 2] = 0;          // tie counter (reset every call)
        tsel[p + 3] = allsel_s;
    }
}

// ---------------------------------------------------------------------------
// K5: tiers + sel_key = keep_score - tier*1e6
// ---------------------------------------------------------------------------
__global__ __launch_bounds__(256) void tier_mark(const float* __restrict__ lsc,
                                                 const float* __restrict__ psc,
                                                 const float* __restrict__ keepS,
                                                 int* __restrict__ tsel,
                                                 float* __restrict__ selk) {
    int gid = blockIdx.x * 256 + threadIdx.x;
    if (gid >= BB * NITEMS) return;
    int b = gid / NITEMS, item = gid % NITEMS;
    int s; bool valid;
    if (item < NL)           { s = 0; valid = lsc[b * NL + item] > 0.f; }
    else if (item < NL + NP) { s = 1; valid = psc[b * NP + (item - NL)] > 0.f; }
    else                     { s = 2; valid = true; }
    float ks = keepS[gid];
    int tier;
    if (!valid) tier = 2;
    else {
        int p = (b * 3 + s) * 4;
        unsigned key = f2key(ks);
        unsigned T = (unsigned)tsel[p];
        int need = tsel[p + 1];
        if (key > T) tier = 0;
        else if (key == T && need > 0 && atomicAdd(&tsel[p + 2], 1) < need) tier = 0;
        else tier = 1;
    }
    selk[gid] = ks - (float)tier * 1000000.0f;
}

// ---------------------------------------------------------------------------
// K6: per-batch radix-select of 900th largest sel_key
// ---------------------------------------------------------------------------
__global__ __launch_bounds__(256) void batch_select(const float* __restrict__ selk,
                                                    int* __restrict__ bsel) {
    int b = blockIdx.x;
    __shared__ unsigned hist[256];
    __shared__ unsigned prefix_s;
    __shared__ int kk_s, allsel_s;
    if (threadIdx.x == 0) { prefix_s = 0; kk_s = KSEL; allsel_s = 0; }
    __syncthreads();
    for (int byt = 3; byt >= 0; --byt) {
        hist[threadIdx.x] = 0;
        __syncthreads();
        unsigned pre = prefix_s;
        for (int i = threadIdx.x; i < NITEMS; i += 256) {
            unsigned key = f2key(selk[b * NITEMS + i]);
            if (byt == 3 || ((key ^ pre) >> ((byt + 1) * 8)) == 0)
                atomicAdd(&hist[(key >> (byt * 8)) & 255], 1u);
        }
        __syncthreads();
        if (threadIdx.x == 0) {
            int kk = kk_s, found = -1;
            for (int bin = 255; bin >= 0; --bin) {
                int c = (int)hist[bin];
                if (c >= kk) { found = bin; break; }
                kk -= c;
            }
            if (found < 0) allsel_s = 1;
            else { prefix_s = pre | ((unsigned)found << (byt * 8)); kk_s = kk; }
        }
        __syncthreads();
        if (allsel_s) break;
    }
    if (threadIdx.x == 0) {
        int p = b * 4;
        if (allsel_s) { bsel[p] = 0; bsel[p + 1] = 0; }
        else          { bsel[p] = (int)prefix_s; bsel[p + 1] = kk_s; }
        bsel[p + 2] = 0;   // tie counter
        bsel[p + 3] = 0;   // selection counter
    }
}

// ---------------------------------------------------------------------------
// K7: compact the 900 winners (unordered)
// ---------------------------------------------------------------------------
__global__ __launch_bounds__(256) void compact_sel(const float* __restrict__ selk,
                                                   int* __restrict__ bsel,
                                                   int* __restrict__ selIdx) {
    int gid = blockIdx.x * 256 + threadIdx.x;
    if (gid >= BB * NITEMS) return;
    int b = gid / NITEMS, item = gid % NITEMS;
    int p = b * 4;
    unsigned key = f2key(selk[gid]);
    unsigned T = (unsigned)bsel[p];
    int need = bsel[p + 1];
    bool take = false;
    if (key > T) take = true;
    else if (key == T && need > 0 && atomicAdd(&bsel[p + 2], 1) < need) take = true;
    if (take) {
        int pos = atomicAdd(&bsel[p + 3], 1);
        if (pos < 1024) selIdx[b * 1024 + pos] = item;
    }
}

// ---------------------------------------------------------------------------
// K8: bitonic sort of the 900 selected by keep_score desc (tie: index asc)
// ---------------------------------------------------------------------------
__global__ __launch_bounds__(1024) void sort_sel(const float* __restrict__ keepS,
                                                 int* __restrict__ selIdx) {
    int b = blockIdx.x, tid = threadIdx.x;
    __shared__ unsigned long long key[1024];
    unsigned long long kv = 0ull;
    if (tid < KSEL) {
        int idx = selIdx[b * 1024 + tid];
        float sc = keepS[b * NITEMS + idx];
        kv = ((unsigned long long)f2key(sc) << 32) |
             (unsigned long long)(0xFFFFFFFFu - (unsigned)idx);
    }
    key[tid] = kv;
    __syncthreads();
    for (int size = 2; size <= 1024; size <<= 1)
        for (int stride = size >> 1; stride > 0; stride >>= 1) {
            int ixj = tid ^ stride;
            if (ixj > tid) {
                bool up = ((tid & size) == 0);      // ascending network
                unsigned long long a = key[tid], c = key[ixj];
                if ((a > c) == up) { key[tid] = c; key[ixj] = a; }
            }
            __syncthreads();
        }
    if (tid < KSEL) {
        unsigned long long v = key[1023 - tid];     // descending ranks
        selIdx[b * 1024 + tid] = (int)(0xFFFFFFFFu - (unsigned)(v & 0xFFFFFFFFu));
    }
}

// ---------------------------------------------------------------------------
// K9: gather outputs
// ---------------------------------------------------------------------------
__global__ __launch_bounds__(256) void gather_out(
        const float* __restrict__ lq, const float* __restrict__ pq,
        const float* __restrict__ gqv,
        const float* __restrict__ lr, const float* __restrict__ pr,
        const float* __restrict__ grf,
        const float* __restrict__ se, const float* __restrict__ scores,
        const float* __restrict__ keepL, const int* __restrict__ selIdx,
        float* __restrict__ out) {
    int b = blockIdx.x / KSEL, j = blockIdx.x % KSEL;
    int idx = selIdx[b * 1024 + j];
    int s; const float* q; const float* rf;
    if (idx < NL)           { s = 0; q = lq + ((size_t)b * NL + idx) * DD;
                              rf = lr + ((size_t)b * NL + idx) * 3; }
    else if (idx < NL + NP) { s = 1; int li = idx - NL;
                              q = pq + ((size_t)b * NP + li) * DD;
                              rf = pr + ((size_t)b * NP + li) * 3; }
    else                    { s = 2; int li = idx - NL - NP;
                              q = gqv + ((size_t)b * NG + li) * DD;
                              rf = grf + ((size_t)b * NG + li) * 3; }
    int d = threadIdx.x;
    size_t row = (size_t)b * KSEL + j;
    out[O_GQ + row * DD + d] = q[d] + se[s * DD + d];
    if (d < 3) out[O_GR + row * 3 + d] = rf[d];
    if (d == 0) {
        out[O_GS  + row] = scores[b * NITEMS + idx];
        out[O_SRC + row] = (float)s;
        out[O_GL  + row] = keepL[b * NITEMS + idx];
    }
}

// ---------------------------------------------------------------------------
extern "C" void kernel_launch(void* const* d_in, const int* in_sizes, int n_in,
                              void* d_out, int out_size, void* d_ws, size_t ws_size,
                              hipStream_t stream) {
    (void)in_sizes; (void)n_in; (void)out_size; (void)ws_size;
    const float* lq  = (const float*)d_in[0];
    const float* lr  = (const float*)d_in[1];
    const float* lsc = (const float*)d_in[2];
    const float* pq  = (const float*)d_in[3];
    const float* pr  = (const float*)d_in[4];
    const float* psc = (const float*)d_in[5];
    const float* gq  = (const float*)d_in[6];
    const float* grf = (const float*)d_in[7];
    const float* gsc = (const float*)d_in[8];
    const float* se  = (const float*)d_in[9];
    const float* w1  = (const float*)d_in[10];
    const float* b1  = (const float*)d_in[11];
    const float* w2  = (const float*)d_in[12];
    const float* b2  = (const float*)d_in[13];
    float* out = (float*)d_out;

    char* ws = (char*)d_ws;
    _Float16* w1t = (_Float16*)(ws);                 // 131072 B
    float* scores = (float*)(ws + 131072);           // 736256 B
    float* keepL  = (float*)(ws + 867328);           // 736256 B
    float* keepS  = (float*)(ws + 1603584);          // 736256 B
    float* selk   = (float*)(ws + 2339840);          // 736256 B
    float* normP  = (float*)(ws + 3076096);          // 512 B
    int*   tsel   = (int*)(ws + 3076608);            // 512 B
    int*   bsel   = (int*)(ws + 3077120);            // 256 B
    int*   selIdx = (int*)(ws + 3077376);            // 32768 B

    int ew = (BB * NITEMS + 255) / 256;              // elementwise grid

    conv_w1<<<256, 256, 0, stream>>>(w1, w1t);
    norm_reduce<<<24, 256, 0, stream>>>(lsc, psc, gsc, normP);
    norm_apply<<<ew, 256, 0, stream>>>(lsc, psc, gsc, normP, scores);
    mlp_wmma<<<MLP_BLOCKS, 256, LDS_BYTES, stream>>>(
        lq, pq, gq, se, w1, b1, w2, b2, w1t, scores, keepL, keepS);
    tier_select<<<24, 256, 0, stream>>>(lsc, psc, keepS, tsel);
    tier_mark<<<ew, 256, 0, stream>>>(lsc, psc, keepS, tsel, selk);
    batch_select<<<BB, 256, 0, stream>>>(selk, bsel);
    compact_sel<<<ew, 256, 0, stream>>>(selk, bsel, selIdx);
    sort_sel<<<BB, 1024, 0, stream>>>(keepS, selIdx);
    gather_out<<<BB * KSEL, 256, 0, stream>>>(lq, pq, gq, lr, pr, grf, se,
                                              scores, keepL, selIdx, out);
}
